// AutoLSTM_68350109548888
// MI455X (gfx1250) — compile-verified, matrix operand order checked by reading, not run on hardware
//
#include <hip/hip_runtime.h>
#include <math.h>

// ---------------------------------------------------------------------------
// MI455X / gfx1250, wave32. Matrix work via v_wmma_f32_16x16x32_bf16.
// ---------------------------------------------------------------------------

typedef __attribute__((ext_vector_type(16))) __bf16 v16bf;
typedef __attribute__((ext_vector_type(8)))  __bf16 v8bf;
typedef __attribute__((ext_vector_type(8)))  float  v8f;

constexpr int NB  = 256;      // batch
constexpr int NT  = 256;      // time
constexpr int BT  = NB * NT;  // 65536
constexpr int HID = 128;      // LSTM hidden
constexpr float SLOPE = 0.01f; // leaky_relu default slope

__device__ __forceinline__ v8f wmma_bf16(v16bf a, v16bf b, v8f c) {
  return __builtin_amdgcn_wmma_f32_16x16x32_bf16(
      /*neg_a=*/false, a, /*neg_b=*/false, b,
      /*c_mod=*/(short)0, c, /*reuse_a=*/false, /*reuse_b=*/false);
}

// Load a 16x32 bf16 fragment (A or B^T) for one lane from a row-major row.
// Lane layout (ISA 7.12.2): lanes 0-15 hold K {0..7,16..23}, lanes 16-31 hold
// K {8..15,24..31} => koff = (lane>=16)?8:0, two contiguous 8-elem chunks.
__device__ __forceinline__ v16bf frag_ld(const __bf16* p, int koff) {
  v8bf lo = *(const v8bf*)(p + koff);
  v8bf hi = *(const v8bf*)(p + koff + 16);
  v16bf r;
#pragma unroll
  for (int i = 0; i < 8; ++i) { r[i] = lo[i]; r[i + 8] = hi[i]; }
  return r;
}

// Same, but converting from f32 source on the fly (used for LSTM h).
__device__ __forceinline__ v16bf frag_ld_f32(const float* p, int koff) {
  v16bf r;
#pragma unroll
  for (int i = 0; i < 8; ++i) {
    r[i]     = (__bf16)p[koff + i];
    r[i + 8] = (__bf16)p[koff + 16 + i];
  }
  return r;
}

__device__ __forceinline__ float leaky(float v) { return v > 0.f ? v : SLOPE * v; }
__device__ __forceinline__ float sigm(float v)  { return 1.f / (1.f + expf(-v)); }

// ---------------------------------------------------------------------------
// Small utility kernels
// ---------------------------------------------------------------------------

// Wt[n*K + k] = (bf16) W[k*N + n]   (pack weights as [N,K] for B-fragments)
__global__ void k_transpose_to_bf16(const float* __restrict__ W,
                                    __bf16* __restrict__ Wt, int K, int N) {
  int gid = blockIdx.x * blockDim.x + threadIdx.x;
  if (gid >= K * N) return;
  int k = gid % K, n = gid / K;
  Wt[(unsigned)n * K + k] = (__bf16)W[(unsigned)k * N + n];
}

__global__ void k_f32_to_bf16(const float* __restrict__ s,
                              __bf16* __restrict__ d, int n) {
  int gid = blockIdx.x * blockDim.x + threadIdx.x;
  if (gid < n) d[gid] = (__bf16)s[gid];
}

__global__ void k_zero_f32(float* __restrict__ p, int n) {
  int gid = blockIdx.x * blockDim.x + threadIdx.x;
  if (gid < n) p[gid] = 0.f;
}

// conv1: Cin=8 (too narrow for WMMA k-blocks) -> plain VALU, bf16 output.
__global__ void k_conv1(const float* __restrict__ x, const float* __restrict__ W,
                        const float* __restrict__ b, __bf16* __restrict__ out) {
  int gid = blockIdx.x * blockDim.x + threadIdx.x;   // BT*32 threads
  if (gid >= BT * 32) return;
  int pos = gid >> 5, ch = gid & 31;
  int t = pos & (NT - 1), bbase = pos - t;
  float acc = b[ch];
#pragma unroll
  for (int kk = 0; kk < 5; ++kk) {
    int st = t + kk - 2;
    if ((unsigned)st < (unsigned)NT) {
      const float* xr = x + (unsigned)(bbase + st) * 8;
      const float* wr = W + kk * 8 * 32 + ch;
#pragma unroll
      for (int c = 0; c < 8; ++c) acc += xr[c] * wr[c * 32];
    }
  }
  out[(unsigned)pos * 32 + ch] = (__bf16)leaky(acc);
}

// ---------------------------------------------------------------------------
// Fused im2col conv GEMM (K=5, SAME): Out[BT,Cout] = im2col(Ain) @ Wt^T
// Ain bf16 [BT,CIN], Wt bf16 [COUT, 5*CIN], Out bf16 with leaky-relu.
// CIN % 32 == 0 so each 32-wide k-block stays inside one conv tap.
// Loop order: tap (padding predicate hoisted) -> channel-block; invalid taps
// contribute zero and are skipped entirely (loads + WMMAs).
// Wave tile: 16 rows x 64 cols (4 f32 accumulators). Block = 8 waves.
// ---------------------------------------------------------------------------
template <int CIN, int COUT>
__global__ __launch_bounds__(256)
void k_conv_gemm(const __bf16* __restrict__ Ain, const __bf16* __restrict__ Wt,
                 const float* __restrict__ bias, __bf16* __restrict__ Out) {
  constexpr int KF = 5 * CIN;
  constexpr int CG = COUT / 64;
  int wid  = (blockIdx.x * blockDim.x + threadIdx.x) >> 5;
  int lane = threadIdx.x & 31;
  int mt = wid / CG, g = wid % CG;
  int r0 = mt << 4, n0 = g << 6;
  int lrow = lane & 15;
  int row  = r0 + lrow;
  int t = row & (NT - 1), bbase = row - t;
  int koff = (lane >= 16) ? 8 : 0;
  int rtop = (lane >= 16) ? 8 : 0;

  v8f acc[4];
  const __bf16* bp[4];
#pragma unroll
  for (int j = 0; j < 4; ++j) {
    int n = n0 + j * 16 + lrow;
    bp[j] = Wt + (unsigned)n * KF;
    float bv = bias[n];
#pragma unroll
    for (int i = 0; i < 8; ++i) acc[j][i] = bv;
  }

#pragma unroll
  for (int kk = 0; kk < 5; ++kk) {
    int st = t + kk - 2;                       // SAME padding (2 each side)
    if ((unsigned)st < (unsigned)NT) {
      const __bf16* arow = Ain + (unsigned)((bbase + st) * CIN);
      const int kbase = kk * CIN;
      for (int cb = 0; cb < CIN; cb += 32) {
        // hoist all fragment loads ahead of the WMMA chain
        v16bf af = frag_ld(arow + cb, koff);
        v16bf b0 = frag_ld(bp[0] + kbase + cb, koff);
        v16bf b1 = frag_ld(bp[1] + kbase + cb, koff);
        v16bf b2 = frag_ld(bp[2] + kbase + cb, koff);
        v16bf b3 = frag_ld(bp[3] + kbase + cb, koff);
        acc[0] = wmma_bf16(af, b0, acc[0]);
        acc[1] = wmma_bf16(af, b1, acc[1]);
        acc[2] = wmma_bf16(af, b2, acc[2]);
        acc[3] = wmma_bf16(af, b3, acc[3]);
      }
    }
  }
#pragma unroll
  for (int j = 0; j < 4; ++j) {
    int n = n0 + j * 16 + lrow;
#pragma unroll
    for (int i = 0; i < 8; ++i)
      Out[(unsigned)(r0 + rtop + i) * COUT + n] = (__bf16)leaky(acc[j][i]);
  }
}

// ---------------------------------------------------------------------------
// Generic GEMM: C[M,N] (f32) = A[M,K] (bf16) @ Wt[N,K]^T (bf16) + bias
// K=128, N=512 instances fully unroll (4 k-steps, immediate offsets).
// ---------------------------------------------------------------------------
template <int K, int N>
__global__ __launch_bounds__(256)
void k_gemm(const __bf16* __restrict__ A, const __bf16* __restrict__ Wt,
            const float* __restrict__ bias, float* __restrict__ C) {
  constexpr int CG = N / 64;
  int wid  = (blockIdx.x * blockDim.x + threadIdx.x) >> 5;
  int lane = threadIdx.x & 31;
  int mt = wid / CG, g = wid % CG;
  int r0 = mt << 4, n0 = g << 6;
  int lrow = lane & 15;
  int koff = (lane >= 16) ? 8 : 0;
  int rtop = (lane >= 16) ? 8 : 0;

  v8f acc[4];
  const __bf16* bp[4];
#pragma unroll
  for (int j = 0; j < 4; ++j) {
    int n = n0 + j * 16 + lrow;
    bp[j] = Wt + (unsigned)n * K;
    float bv = bias[n];
#pragma unroll
    for (int i = 0; i < 8; ++i) acc[j][i] = bv;
  }
  const __bf16* arow = A + (unsigned)(r0 + lrow) * K;
#pragma unroll 4
  for (int kb = 0; kb < K / 32; ++kb) {
    int ks = kb << 5;
    v16bf af = frag_ld(arow + ks, koff);
    v16bf b0 = frag_ld(bp[0] + ks, koff);
    v16bf b1 = frag_ld(bp[1] + ks, koff);
    v16bf b2 = frag_ld(bp[2] + ks, koff);
    v16bf b3 = frag_ld(bp[3] + ks, koff);
    acc[0] = wmma_bf16(af, b0, acc[0]);
    acc[1] = wmma_bf16(af, b1, acc[1]);
    acc[2] = wmma_bf16(af, b2, acc[2]);
    acc[3] = wmma_bf16(af, b3, acc[3]);
  }
#pragma unroll
  for (int j = 0; j < 4; ++j) {
    int n = n0 + j * 16 + lrow;
#pragma unroll
    for (int i = 0; i < 8; ++i)
      C[(unsigned)(r0 + rtop + i) * N + n] = acc[j][i];
  }
}

// ---------------------------------------------------------------------------
// Split-K GEMM for the dense layer (M=256, N=512, K=32768).
// Without split-K only 128 waves exist -> latency-bound. PARTS K-partitions
// run in parallel (grid.y), each accumulating f32 partials into its own
// buffer slice; a deterministic reduction adds partials + bias (no float
// atomics -> bit-stable across graph replays).
// ---------------------------------------------------------------------------
template <int K, int N, int PARTS>
__global__ __launch_bounds__(256)
void k_gemm_splitk(const __bf16* __restrict__ A, const __bf16* __restrict__ Wt,
                   float* __restrict__ Cpart) {
  constexpr int KP = K / PARTS;
  constexpr int CG = N / 64;
  int part = blockIdx.y;
  int wid  = (blockIdx.x * blockDim.x + threadIdx.x) >> 5;
  int lane = threadIdx.x & 31;
  int mt = wid / CG, g = wid % CG;
  int r0 = mt << 4, n0 = g << 6;
  int lrow = lane & 15;
  int koff = (lane >= 16) ? 8 : 0;
  int rtop = (lane >= 16) ? 8 : 0;

  v8f acc[4];
  const __bf16* bp[4];
#pragma unroll
  for (int j = 0; j < 4; ++j) {
    int n = n0 + j * 16 + lrow;
    bp[j] = Wt + (unsigned)n * K + (unsigned)part * KP;
#pragma unroll
    for (int i = 0; i < 8; ++i) acc[j][i] = 0.f;
  }
  const __bf16* arow = A + (unsigned)(r0 + lrow) * K + (unsigned)part * KP;
#pragma unroll 4
  for (int kb = 0; kb < KP / 32; ++kb) {
    int ks = kb << 5;
    v16bf af = frag_ld(arow + ks, koff);
    v16bf b0 = frag_ld(bp[0] + ks, koff);
    v16bf b1 = frag_ld(bp[1] + ks, koff);
    v16bf b2 = frag_ld(bp[2] + ks, koff);
    v16bf b3 = frag_ld(bp[3] + ks, koff);
    acc[0] = wmma_bf16(af, b0, acc[0]);
    acc[1] = wmma_bf16(af, b1, acc[1]);
    acc[2] = wmma_bf16(af, b2, acc[2]);
    acc[3] = wmma_bf16(af, b3, acc[3]);
  }
  float* cp = Cpart + (unsigned)part * (NB * N);
#pragma unroll
  for (int j = 0; j < 4; ++j) {
    int n = n0 + j * 16 + lrow;
#pragma unroll
    for (int i = 0; i < 8; ++i)
      cp[(unsigned)(r0 + rtop + i) * N + n] = acc[j][i];
  }
}

// out[m,n] = bias[n] + sum_p Cpart[p][m][n]   (256x512 elements)
template <int N, int PARTS>
__global__ void k_reduce_bias(const float* __restrict__ Cpart,
                              const float* __restrict__ bias,
                              float* __restrict__ out) {
  int gid = blockIdx.x * blockDim.x + threadIdx.x;
  if (gid >= NB * N) return;
  float v = bias[gid & (N - 1)];
#pragma unroll
  for (int p = 0; p < PARTS; ++p) v += Cpart[(unsigned)p * (NB * N) + gid];
  out[gid] = v;
}

// ---------------------------------------------------------------------------
// One LSTM time step (fused): z = xpart[:,t,:] + h @ Wth^T ; gates ; update.
// Grid: 16 blocks (16 batch rows each) x 256 threads (8 waves x 4 col-tiles).
// Each block touches only its own 16 rows of h/c -> in-place update is safe.
// ---------------------------------------------------------------------------
__global__ __launch_bounds__(256)
void k_lstm_step(const float* __restrict__ xpart, const __bf16* __restrict__ Wth,
                 float* __restrict__ cbuf, float* __restrict__ hbuf,
                 float* __restrict__ outseq, int t) {
  __shared__ float zs[16 * 512];           // 32 KB of 320 KB WGP LDS
  int lane = threadIdx.x & 31;
  int w    = threadIdx.x >> 5;             // wave 0..7
  int m0   = blockIdx.x << 4;              // batch-row base
  int lrow = lane & 15;
  int koff = (lane >= 16) ? 8 : 0;
  int rtop = (lane >= 16) ? 8 : 0;
  int n0   = w << 6;                       // 64 cols per wave

  v8f acc[4];
  const __bf16* bp[4];
#pragma unroll
  for (int j = 0; j < 4; ++j) {
    int n = n0 + j * 16 + lrow;
    bp[j] = Wth + (unsigned)n * HID;
#pragma unroll
    for (int i = 0; i < 8; ++i)
      acc[j][i] = xpart[(unsigned)((m0 + rtop + i) * NT + t) * 512 + n];
  }
  const float* hrow = hbuf + (unsigned)(m0 + lrow) * HID;
#pragma unroll
  for (int kb = 0; kb < 4; ++kb) {         // K = 128
    int ks = kb << 5;
    v16bf af = frag_ld_f32(hrow + ks, koff);
    v16bf b0 = frag_ld(bp[0] + ks, koff);
    v16bf b1 = frag_ld(bp[1] + ks, koff);
    v16bf b2 = frag_ld(bp[2] + ks, koff);
    v16bf b3 = frag_ld(bp[3] + ks, koff);
    acc[0] = wmma_bf16(af, b0, acc[0]);
    acc[1] = wmma_bf16(af, b1, acc[1]);
    acc[2] = wmma_bf16(af, b2, acc[2]);
    acc[3] = wmma_bf16(af, b3, acc[3]);
  }
#pragma unroll
  for (int j = 0; j < 4; ++j) {
    int n = n0 + j * 16 + lrow;
#pragma unroll
    for (int i = 0; i < 8; ++i) zs[(rtop + i) * 512 + n] = acc[j][i];
  }
  __syncthreads();

  // gates: 16 rows x 128 hidden = 2048 values, 8 per thread
#pragma unroll
  for (int e = 0; e < 8; ++e) {
    int f  = threadIdx.x * 8 + e;
    int lr = f >> 7, hh = f & 127;
    float zi = zs[lr * 512 + hh];
    float zf = zs[lr * 512 + 128 + hh];
    float zg = zs[lr * 512 + 256 + hh];
    float zo = zs[lr * 512 + 384 + hh];
    int m = m0 + lr;
    float cold = cbuf[(unsigned)m * HID + hh];
    float cn = sigm(zf) * cold + sigm(zi) * tanhf(zg);
    float hn = sigm(zo) * tanhf(cn);
    cbuf[(unsigned)m * HID + hh] = cn;
    hbuf[(unsigned)m * HID + hh] = hn;
    outseq[(unsigned)(m * NT + t) * HID + hh] = hn;
  }
}

// ---------------------------------------------------------------------------
// BatchNorm over the batch axis (population stats) + leaky relu, in place.
// One block per column (512 blocks x 256 threads).
// ---------------------------------------------------------------------------
__global__ void k_bn(float* __restrict__ d, const float* __restrict__ scale,
                     const float* __restrict__ bias) {
  __shared__ float s1[256], s2[256];
  int col = blockIdx.x, r = threadIdx.x;
  float v = d[(unsigned)r * 512 + col];
  s1[r] = v; s2[r] = v * v;
  __syncthreads();
  for (int s = 128; s > 0; s >>= 1) {
    if (r < s) { s1[r] += s1[r + s]; s2[r] += s2[r + s]; }
    __syncthreads();
  }
  float mean = s1[0] * (1.f / 256.f);
  float var  = s2[0] * (1.f / 256.f) - mean * mean;
  float y = (v - mean) * rsqrtf(var + 1e-5f) * scale[col] + bias[col];
  d[(unsigned)r * 512 + col] = leaky(y);
}

// Final 512->10 matvec + softmax. One wave per batch row.
__global__ void k_head(const float* __restrict__ d, const float* __restrict__ W,
                       const float* __restrict__ b, float* __restrict__ out) {
  __shared__ float lg[10];
  int row = blockIdx.x, lane = threadIdx.x;
  if (lane < 10) {
    float acc = b[lane];
    for (int k = 0; k < 512; ++k) acc += d[(unsigned)row * 512 + k] * W[k * 10 + lane];
    lg[lane] = acc;
  }
  __syncthreads();
  if (lane < 10) {
    float m = lg[0];
#pragma unroll
    for (int j = 1; j < 10; ++j) m = fmaxf(m, lg[j]);
    float s = 0.f;
#pragma unroll
    for (int j = 0; j < 10; ++j) s += expf(lg[j] - m);
    out[row * 10 + lane] = expf(lg[lane] - m) / s;
  }
}

// ---------------------------------------------------------------------------
// Host-side orchestration
// ---------------------------------------------------------------------------
extern "C" void kernel_launch(void* const* d_in, const int* in_sizes, int n_in,
                              void* d_out, int out_size, void* d_ws, size_t ws_size,
                              hipStream_t stream) {
  (void)in_sizes; (void)n_in; (void)out_size; (void)ws_size;
  const float* x    = (const float*)d_in[0];
  const float* cW1  = (const float*)d_in[1];
  const float* cb1  = (const float*)d_in[2];
  const float* cW2  = (const float*)d_in[3];
  const float* cb2  = (const float*)d_in[4];
  const float* cW3  = (const float*)d_in[5];
  const float* cb3  = (const float*)d_in[6];
  const float* Wx1  = (const float*)d_in[7];
  const float* Wh1  = (const float*)d_in[8];
  const float* b1   = (const float*)d_in[9];
  const float* Wx2  = (const float*)d_in[10];
  const float* Wh2  = (const float*)d_in[11];
  const float* b2   = (const float*)d_in[12];
  const float* Wd1  = (const float*)d_in[13];
  const float* bd1  = (const float*)d_in[14];
  const float* bns  = (const float*)d_in[15];
  const float* bnb  = (const float*)d_in[16];
  const float* Wd2  = (const float*)d_in[17];
  const float* bd2  = (const float*)d_in[18];

  char* ws = (char*)d_ws;
  size_t off = 0;
  auto take = [&](size_t bytes) -> char* {
    char* p = ws + off;
    off += (bytes + 255) & ~(size_t)255;
    return p;
  };

  __bf16* Wt2   = (__bf16*)take((size_t)512 * 160 * 2);
  __bf16* Wt3   = (__bf16*)take((size_t)128 * 2560 * 2);
  __bf16* Wtx1  = (__bf16*)take((size_t)512 * 128 * 2);
  __bf16* Wth1  = (__bf16*)take((size_t)512 * 128 * 2);
  __bf16* Wtx2  = (__bf16*)take((size_t)512 * 128 * 2);
  __bf16* Wth2  = (__bf16*)take((size_t)512 * 128 * 2);
  __bf16* Wtd1  = (__bf16*)take((size_t)512 * 32768 * 2);
  __bf16* a1    = (__bf16*)take((size_t)BT * 32 * 2);
  __bf16* a2    = (__bf16*)take((size_t)BT * 512 * 2);
  __bf16* a3    = (__bf16*)take((size_t)BT * 128 * 2);
  float*  xpart = (float*) take((size_t)BT * 512 * 4);   // reused by both layers
  float*  out1f = (float*) take((size_t)BT * 128 * 4);
  __bf16* out1b = (__bf16*)take((size_t)BT * 128 * 2);
  float*  out2f = (float*) take((size_t)BT * 128 * 4);
  __bf16* out2b = (__bf16*)take((size_t)BT * 128 * 2);
  float*  hbuf  = (float*) take((size_t)NB * HID * 4);
  float*  cbuf  = (float*) take((size_t)NB * HID * 4);
  float*  dmat  = (float*) take((size_t)NB * 512 * 4);
  float*  cpart = (float*) take((size_t)8 * NB * 512 * 4);  // split-K partials

  auto blocks = [](int n) { return (n + 255) / 256; };

  // Weight packing (f32 [K,N] -> bf16 [N,K])
  k_transpose_to_bf16<<<blocks(160 * 512), 256, 0, stream>>>(cW2, Wt2, 160, 512);
  k_transpose_to_bf16<<<blocks(2560 * 128), 256, 0, stream>>>(cW3, Wt3, 2560, 128);
  k_transpose_to_bf16<<<blocks(128 * 512), 256, 0, stream>>>(Wx1, Wtx1, 128, 512);
  k_transpose_to_bf16<<<blocks(128 * 512), 256, 0, stream>>>(Wh1, Wth1, 128, 512);
  k_transpose_to_bf16<<<blocks(128 * 512), 256, 0, stream>>>(Wx2, Wtx2, 128, 512);
  k_transpose_to_bf16<<<blocks(128 * 512), 256, 0, stream>>>(Wh2, Wth2, 128, 512);
  k_transpose_to_bf16<<<blocks(32768 * 512), 256, 0, stream>>>(Wd1, Wtd1, 32768, 512);

  // Conv stack
  k_conv1<<<blocks(BT * 32), 256, 0, stream>>>(x, cW1, cb1, a1);
  k_conv_gemm<32, 512><<<(BT / 16) * (512 / 64) / 8, 256, 0, stream>>>(a1, Wt2, cb2, a2);
  k_conv_gemm<512, 128><<<(BT / 16) * (128 / 64) / 8, 256, 0, stream>>>(a2, Wt3, cb3, a3);

  // LSTM layer 1: hoisted x@Wx1 + b1, then 256 fused recurrent steps
  k_gemm<128, 512><<<(BT / 16) * (512 / 64) / 8, 256, 0, stream>>>(a3, Wtx1, b1, xpart);
  k_zero_f32<<<blocks(NB * HID), 256, 0, stream>>>(hbuf, NB * HID);
  k_zero_f32<<<blocks(NB * HID), 256, 0, stream>>>(cbuf, NB * HID);
  for (int t = 0; t < NT; ++t)
    k_lstm_step<<<NB / 16, 256, 0, stream>>>(xpart, Wth1, cbuf, hbuf, out1f, t);

  // LSTM layer 2 (carry inherited from layer 1 — do NOT re-zero h/c)
  k_f32_to_bf16<<<blocks(BT * 128), 256, 0, stream>>>(out1f, out1b, BT * 128);
  k_gemm<128, 512><<<(BT / 16) * (512 / 64) / 8, 256, 0, stream>>>(out1b, Wtx2, b2, xpart);
  for (int t = 0; t < NT; ++t)
    k_lstm_step<<<NB / 16, 256, 0, stream>>>(xpart, Wth2, cbuf, hbuf, out2f, t);

  // Dense [256, 32768] @ [32768, 512]: deterministic split-K (8 partitions,
  // 128 blocks instead of 16) + reduction with bias; then BN + head.
  k_f32_to_bf16<<<blocks(BT * 128), 256, 0, stream>>>(out2f, out2b, BT * 128);
  {
    dim3 grid((NB / 16) * (512 / 64) / 8, 8);
    k_gemm_splitk<32768, 512, 8><<<grid, 256, 0, stream>>>(out2b, Wtd1, cpart);
  }
  k_reduce_bias<512, 8><<<blocks(NB * 512), 256, 0, stream>>>(cpart, bd1, dmat);
  k_bn<<<512, 256, 0, stream>>>(dmat, bns, bnb);
  k_head<<<NB, 32, 0, stream>>>(dmat, Wd2, bd2, (float*)d_out);
}
